// Model_57612691309114
// MI455X (gfx1250) — matrix operand order repaired and checked by reading, not compiled, not run on hardware
//
#include <hip/hip_runtime.h>
#include <math.h>

#define DEV_INLINE __device__ __forceinline__

namespace {
constexpr int V = 50000;
constexpr int E = 5000;
constexpr int C = 16;
constexpr float TAU = 0.5f;
constexpr float EPS = 1e-8f;
constexpr int NBLK  = (V + 31) / 32;   // 1563 vertex blocks of 32
constexpr int NTILE = (E + 15) / 16;   // 313 edge tiles of 16
constexpr int NWAVE = 8;               // 256 threads = 8 wave32

// workspace layout in dwords
constexpr size_t PS_F  = 0;                            // [V*C] f32 softmax(pred_s)
constexpr size_t PT_F  = PS_F + (size_t)V * C;         // [V*C] f32 softmax(pred_t)
constexpr size_t BPACK = (size_t)NBLK * 256;           // per pre-swizzled array
constexpr size_t BS_HI = PT_F + (size_t)V * C;
constexpr size_t BS_LO = BS_HI + BPACK;
constexpr size_t BT_HI = BS_LO + BPACK;
constexpr size_t BT_LO = BT_HI + BPACK;
constexpr size_t PARTIAL = BT_LO + BPACK;              // [NTILE] f32
}  // namespace

typedef __attribute__((ext_vector_type(16))) __bf16        v16bf;
typedef __attribute__((ext_vector_type(8)))  float         v8f;
typedef __attribute__((ext_vector_type(8)))  unsigned int  v8u;
typedef __attribute__((ext_vector_type(4)))  unsigned int  v4u;

DEV_INLINE unsigned bf16_rne(float x) {
  unsigned u = __float_as_uint(x);
  return (u + 0x7FFFu + ((u >> 16) & 1u)) >> 16;
}
DEV_INLINE float bf16_to_f32(unsigned h) { return __uint_as_float(h << 16); }

// ---------------------------------------------------------------------------
// Kernel 1: row softmax of pred_s / pred_t (C = 16)
// ---------------------------------------------------------------------------
__global__ void __launch_bounds__(256) softmax_kernel(
    const float* __restrict__ pred_s, const float* __restrict__ pred_t,
    float* __restrict__ ps_f, float* __restrict__ pt_f) {
  int v = blockIdx.x * blockDim.x + threadIdx.x;
  if (v >= V) return;
#pragma unroll
  for (int which = 0; which < 2; ++which) {
    const float* in = (which == 0 ? pred_s : pred_t) + (size_t)v * C;
    float* out = (which == 0 ? ps_f : pt_f) + (size_t)v * C;
    float x[C];
    float mx = -INFINITY;
#pragma unroll
    for (int i = 0; i < C; ++i) { x[i] = in[i]; mx = fmaxf(mx, x[i]); }
    float sum = 0.f;
#pragma unroll
    for (int i = 0; i < C; ++i) { x[i] = expf(x[i] - mx); sum += x[i]; }
    float inv = 1.0f / sum;
#pragma unroll
    for (int i = 0; i < C; ++i) out[i] = x[i] * inv;
  }
}

// ---------------------------------------------------------------------------
// Kernel 2: bf16 hi/lo split of probabilities, pre-swizzled into the WMMA
// B-operand order used here (16-bit B, K=32, contiguous K, same as A): lane L -> column c = L&15; group g=L>>4;
// VGPR j (dword j) holds K = g*16 + 2j (low 16b) and +1 (high 16b).
// Stored as [blk][lane][j] dwords so a lane fetches its 8 dwords with 2xb128.
// ---------------------------------------------------------------------------
__global__ void __launch_bounds__(256) swizzle_kernel(
    const float* __restrict__ ps_f, const float* __restrict__ pt_f,
    unsigned* __restrict__ bsh, unsigned* __restrict__ bsl,
    unsigned* __restrict__ bth, unsigned* __restrict__ btl) {
  int blk = blockIdx.x;
  int lane = threadIdx.x >> 3;
  int j = threadIdx.x & 7;
  int k0 = (lane >> 4) * 16 + 2 * j;
  int va = blk * 32 + k0;
  int vb = va + 1;
  int c = lane & 15;
  float sa = (va < V) ? ps_f[(size_t)va * C + c] : 0.f;
  float sb = (vb < V) ? ps_f[(size_t)vb * C + c] : 0.f;
  float ta = (va < V) ? pt_f[(size_t)va * C + c] : 0.f;
  float tb = (vb < V) ? pt_f[(size_t)vb * C + c] : 0.f;

  unsigned sh0 = bf16_rne(sa), sh1 = bf16_rne(sb);
  unsigned sl0 = bf16_rne(sa - bf16_to_f32(sh0));
  unsigned sl1 = bf16_rne(sb - bf16_to_f32(sh1));
  unsigned th0 = bf16_rne(ta), th1 = bf16_rne(tb);
  unsigned tl0 = bf16_rne(ta - bf16_to_f32(th0));
  unsigned tl1 = bf16_rne(tb - bf16_to_f32(th1));

  size_t o = (size_t)blk * 256 + (size_t)lane * 8 + j;
  bsh[o] = sh0 | (sh1 << 16);
  bsl[o] = sl0 | (sl1 << 16);
  bth[o] = th0 | (th1 << 16);
  btl[o] = tl0 | (tl1 << 16);
}

// ---------------------------------------------------------------------------
// Kernel 3: fused SpMM (H^T @ ps, H^T @ pt, deg) + per-tile KL partials.
// A-operand (16-bit A, 16x32): lane L -> row e = L&15, group g = L>>4;
// dword j in 0..7 holds K = g*16 + 2j (,+1), same contiguous K order as the B pack.
// H entries are exactly 0.0/1.0 so f32->bf16 is an exact 16-bit truncate.
// ---------------------------------------------------------------------------
template <bool TAIL>
DEV_INLINE v16bf load_A_tile(const float* __restrict__ Hc, int v0, int g) {
  v8u au;
#pragma unroll
  for (int j = 0; j < 8; ++j) {
    int k = g * 16 + 2 * j;
    int va = v0 + k;
    float f0, f1;
    if (TAIL) {
      f0 = (va < V) ? Hc[(size_t)va * E] : 0.0f;
      f1 = (va + 1 < V) ? Hc[(size_t)(va + 1) * E] : 0.0f;
    } else {
      f0 = Hc[(size_t)va * E];
      f1 = Hc[(size_t)(va + 1) * E];
    }
    au[j] = (__float_as_uint(f0) >> 16) | (__float_as_uint(f1) & 0xFFFF0000u);
  }
  return __builtin_bit_cast(v16bf, au);
}

DEV_INLINE v16bf load_B_tile(const unsigned* __restrict__ base, int blk, int lane) {
  const v4u* p = (const v4u*)(base + (size_t)blk * 256 + (size_t)lane * 8);
  v4u lo = p[0];
  v4u hi = p[1];
  v8u r;
  r[0] = lo[0]; r[1] = lo[1]; r[2] = lo[2]; r[3] = lo[3];
  r[4] = hi[0]; r[5] = hi[1]; r[6] = hi[2]; r[7] = hi[3];
  return __builtin_bit_cast(v16bf, r);
}

#define WMMA_BF16(Cacc, A, B) \
  __builtin_amdgcn_wmma_f32_16x16x32_bf16(false, (A), false, (B), (short)0, (Cacc), false, false)

__global__ void __launch_bounds__(256) spmm_kl_kernel(
    const float* __restrict__ H, const unsigned* __restrict__ bsh,
    const unsigned* __restrict__ bsl, const unsigned* __restrict__ bth,
    const unsigned* __restrict__ btl, const int* __restrict__ e_mask,
    float* __restrict__ partial) {
  __shared__ float red[NWAVE][3][8][32];  // 24 KB

  const int wg = blockIdx.x;
  const int wave = threadIdx.x >> 5;
  const int lane = threadIdx.x & 31;
  const int g = lane >> 4;
  const int e0 = wg * 16;
  const int col = e0 + (lane & 15);
  const float* Hc = H + (col < E ? col : E - 1);  // clamp keeps loads in-bounds

  v8f cs = {0.f, 0.f, 0.f, 0.f, 0.f, 0.f, 0.f, 0.f};
  v8f ct = {0.f, 0.f, 0.f, 0.f, 0.f, 0.f, 0.f, 0.f};
  v8f cd = {0.f, 0.f, 0.f, 0.f, 0.f, 0.f, 0.f, 0.f};

  const v8u ones_u = {0x3F803F80u, 0x3F803F80u, 0x3F803F80u, 0x3F803F80u,
                      0x3F803F80u, 0x3F803F80u, 0x3F803F80u, 0x3F803F80u};
  const v16bf bones = __builtin_bit_cast(v16bf, ones_u);

  int blk = wave;
  for (; blk < NBLK - 1; blk += NWAVE) {
    v16bf a  = load_A_tile<false>(Hc, blk * 32, g);
    v16bf b0 = load_B_tile(bsh, blk, lane);
    v16bf b1 = load_B_tile(bsl, blk, lane);
    v16bf b2 = load_B_tile(bth, blk, lane);
    v16bf b3 = load_B_tile(btl, blk, lane);
    cs = WMMA_BF16(cs, a, b0);
    cs = WMMA_BF16(cs, a, b1);
    ct = WMMA_BF16(ct, a, b2);
    ct = WMMA_BF16(ct, a, b3);
    cd = WMMA_BF16(cd, a, bones);
  }
  if (blk == NBLK - 1) {  // last block: 16 valid + 16 padded vertices
    v16bf a  = load_A_tile<true>(Hc, blk * 32, g);
    v16bf b0 = load_B_tile(bsh, blk, lane);
    v16bf b1 = load_B_tile(bsl, blk, lane);
    v16bf b2 = load_B_tile(bth, blk, lane);
    v16bf b3 = load_B_tile(btl, blk, lane);
    cs = WMMA_BF16(cs, a, b0);
    cs = WMMA_BF16(cs, a, b1);
    ct = WMMA_BF16(ct, a, b2);
    ct = WMMA_BF16(ct, a, b3);
    cd = WMMA_BF16(cd, a, bones);
  }

#pragma unroll
  for (int r = 0; r < 8; ++r) {
    red[wave][0][r][lane] = cs[r];
    red[wave][1][r][lane] = ct[r];
    red[wave][2][r][lane] = cd[r];
  }
  __syncthreads();

  if (wave == 0) {
    for (int w = 1; w < NWAVE; ++w) {
#pragma unroll
      for (int r = 0; r < 8; ++r) {
        cs[r] += red[w][0][r][lane];
        ct[r] += red[w][1][r][lane];
        cd[r] += red[w][2][r][lane];
      }
    }
    // C/D layout: lane -> column c = lane&15, rows e_local = g*8 + r
    float kls = 0.f;
    const int erow = e0 + g * 8;
#pragma unroll
    for (int r = 0; r < 8; ++r) {
      int e = erow + r;
      if (e < E && e_mask[e] != 0) {
        float invd = 1.0f / cd[r];             // deg_e replicated per column
        float at = ct[r] * invd * (1.0f / TAU) + EPS;  // exp(log_t)
        float as = cs[r] * invd * (1.0f / TAU) + EPS;  // exp(log_s)
        kls += at * (logf(at) - logf(as));
      }
    }
#pragma unroll
    for (int off = 16; off; off >>= 1) kls += __shfl_xor(kls, off, 32);
    if (lane == 0) partial[wg] = kls;
  }
}

// ---------------------------------------------------------------------------
// Kernel 4: deterministic final reduction (one wave32)
// ---------------------------------------------------------------------------
__global__ void __launch_bounds__(32) finalize_kernel(
    const float* __restrict__ partial, const int* __restrict__ e_mask,
    float* __restrict__ out) {
  int lane = threadIdx.x;
  float s = 0.f;
  for (int i = lane; i < NTILE; i += 32) s += partial[i];
  float cnt = 0.f;
  for (int i = lane; i < E; i += 32) cnt += (e_mask[i] != 0) ? 1.0f : 0.0f;
#pragma unroll
  for (int off = 16; off; off >>= 1) {
    s += __shfl_xor(s, off, 32);
    cnt += __shfl_xor(cnt, off, 32);
  }
  if (lane == 0) out[0] = s / fmaxf(cnt, 1.0f);
}

// ---------------------------------------------------------------------------
extern "C" void kernel_launch(void* const* d_in, const int* in_sizes, int n_in,
                              void* d_out, int out_size, void* d_ws, size_t ws_size,
                              hipStream_t stream) {
  const float* pred_s = (const float*)d_in[0];
  const float* pred_t = (const float*)d_in[1];
  const float* Hm     = (const float*)d_in[2];
  const int*   e_mask = (const int*)d_in[3];

  float* wsf = (float*)d_ws;
  float* ps_f = wsf + PS_F;
  float* pt_f = wsf + PT_F;
  unsigned* bsh = (unsigned*)(wsf + BS_HI);
  unsigned* bsl = (unsigned*)(wsf + BS_LO);
  unsigned* bth = (unsigned*)(wsf + BT_HI);
  unsigned* btl = (unsigned*)(wsf + BT_LO);
  float* partial = wsf + PARTIAL;
  float* out = (float*)d_out;

  softmax_kernel<<<(V + 255) / 256, 256, 0, stream>>>(pred_s, pred_t, ps_f, pt_f);
  swizzle_kernel<<<NBLK, 256, 0, stream>>>(ps_f, pt_f, bsh, bsl, bth, btl);
  spmm_kl_kernel<<<NTILE, 256, 0, stream>>>(Hm, bsh, bsl, bth, btl, e_mask, partial);
  finalize_kernel<<<1, 32, 0, stream>>>(partial, e_mask, out);
}
